// BLSTM_20504173871452
// MI455X (gfx1250) — matrix operand order repaired and checked by reading, not compiled, not run on hardware
//
#include <hip/hip_runtime.h>
#include <hip/hip_bf16.h>
#include <math.h>

typedef __attribute__((ext_vector_type(16))) __bf16 v16bf;
typedef __attribute__((ext_vector_type(8)))  __bf16 v8bf;
typedef __attribute__((ext_vector_type(8)))  float  v8f;

#define TSEQ  512
#define BATCH 64
#define HID   512
#define GATES 2048   // 4*HID
#define NWG_DIR 16   // workgroups per direction in recurrent kernel
#define HSLICE  32   // H columns per workgroup (HID / NWG_DIR)

// ---------------------------------------------------------------- helpers ---

static __device__ inline float sigf(float x) { return 1.0f / (1.0f + expf(-x)); }

// gfx1250 async global->LDS copy, 16 bytes per lane, tracked by ASYNCcnt.
// LDS address = low 32 bits of the generic pointer (flat->LDS truncation).
static __device__ inline void async_g2l_b128(const void* gsrc, void* ldst) {
  uint32_t loff = (uint32_t)(uintptr_t)ldst;
  asm volatile("global_load_async_to_lds_b128 %0, %1, off"
               :: "v"(loff), "v"((uint64_t)(uintptr_t)gsrc)
               : "memory");
}
#define S_WAIT_ASYNC(n) asm volatile("s_wait_asynccnt " #n ::: "memory")

// B fragment (32x16, bf16) from row-major [N][K] weights: lane -> N = l&15,
// elements e -> K = k0 + 16*(l>>4) + e  (one 32-byte contiguous load per lane)
static __device__ inline v16bf load_b_bf16(const __bf16* base, int ld, int n0, int k0, int lane) {
  int n = n0 + (lane & 15);
  int g = lane >> 4;
  const __bf16* p = base + (size_t)n * ld + k0 + 16 * g;
  return *(const v16bf*)p;
}

// B fragment from an LDS chunk laid out as [rows][64 bytes] (K-chunk = 32)
static __device__ inline v16bf load_b_lds(const char* bsh, int nt, int lane) {
  int row = nt * 16 + (lane & 15);
  int g = lane >> 4;
  const __bf16* p = (const __bf16*)(bsh + row * 64 + g * 16);
  v8bf lo = *(const v8bf*)p;
  v8bf hi = *(const v8bf*)(p + 16);
  return __builtin_shufflevector(lo, hi, 0,1,2,3,4,5,6,7,8,9,10,11,12,13,14,15);
}

// B fragment from LDS-resident full B [128 rows][512 K] bf16 (row pitch 1024B)
static __device__ inline v16bf load_b_lds_full(const char* bsh, int nt, int k0, int lane) {
  int row = nt * 16 + (lane & 15);
  int g = lane >> 4;
  const __bf16* p = (const __bf16*)(bsh + (size_t)row * 1024 + (size_t)(k0 + 8 * g) * 2);
  v8bf lo = *(const v8bf*)p;
  v8bf hi = *(const v8bf*)(p + 16);
  return __builtin_shufflevector(lo, hi, 0,1,2,3,4,5,6,7,8,9,10,11,12,13,14,15);
}

// A fragment (16x32, bf16) from row-major [M][K]: lane -> M = l&15,
// element e -> K = k0 + 16*(e/8) + 8*(l>>4) + e%8  (two 16-byte loads)
static __device__ inline v16bf load_a_bf16(const __bf16* base, int ld, int m0, int k0, int lane) {
  int r = m0 + (lane & 15);
  int g = lane >> 4;
  const __bf16* p = base + (size_t)r * ld + k0 + 8 * g;
  v8bf lo = *(const v8bf*)p;
  v8bf hi = *(const v8bf*)(p + 16);
  return __builtin_shufflevector(lo, hi, 0,1,2,3,4,5,6,7,8,9,10,11,12,13,14,15);
}

// A fragment with on-the-fly fp32 -> bf16 conversion (two 32-byte loads)
static __device__ inline v16bf load_a_f32(const float* base, int ld, int m0, int k0, int lane) {
  int r = m0 + (lane & 15);
  int g = lane >> 4;
  const float* p = base + (size_t)r * ld + k0 + 8 * g;
  v8f lo = *(const v8f*)p;
  v8f hi = *(const v8f*)(p + 16);
  v16bf a;
#pragma unroll
  for (int e = 0; e < 8; ++e) { a[e] = (__bf16)lo[e]; a[8 + e] = (__bf16)hi[e]; }
  return a;
}

static __device__ inline v8f wmma_bf16(v16bf a, v16bf b, v8f c) {
  return __builtin_amdgcn_wmma_f32_16x16x32_bf16(false, a, false, b, (short)0, c, false, false);
}

// async-stage one B chunk (128 N-rows x 32 K, bf16 = 8 KB) into LDS
static __device__ inline void issue_b_chunk(char* dst, const __bf16* W, int n0, int k0, int tid) {
#pragma unroll
  for (int i = 0; i < 2; ++i) {
    int p = tid + 256 * i;             // 512 pieces of 16B
    int row = p >> 2, sub = p & 3;
    const char* g = (const char*)(W + (size_t)(n0 + row) * HID + k0) + sub * 16;
    async_g2l_b128(g, dst + row * 64 + sub * 16);
  }
}

// ------------------------------------------------------------ tiny kernels --

__global__ void f2bf_kernel(const float* __restrict__ in, __bf16* __restrict__ out, int n) {
  int i = blockIdx.x * blockDim.x + threadIdx.x;
  if (i < n) out[i] = (__bf16)in[i];
}

__global__ void zero_kernel(int* p, int n) {
  int i = blockIdx.x * blockDim.x + threadIdx.x;
  if (i < n) p[i] = 0;
}

__global__ void hsum_kernel(const float* __restrict__ hf, const float* __restrict__ hb,
                            __bf16* __restrict__ out, int n) {
  int i = blockIdx.x * blockDim.x + threadIdx.x;
  if (i < n) out[i] = (__bf16)(hf[i] + hb[i]);
}

// --------------------------------------------------- dense WMMA GEMM --------
// out[m][n] = sum_k A[m][k]*W[n][k] + bias1[n] (+ bias2[n])
// M = 32768, K = 512, N = nld (2048 for xproj, 512 for linears).
// B double-buffered in LDS via async-to-LDS; wave computes a 16x128 tile.
template <int AF32, int OUTF32>
__global__ __launch_bounds__(256) void gemm_kernel(const void* __restrict__ Aptr,
                                                   const __bf16* __restrict__ W,
                                                   const float* __restrict__ bias1,
                                                   const float* __restrict__ bias2,
                                                   void* __restrict__ outp, int nld) {
  __shared__ __align__(32) char bsh[2][8192];
  int tid = threadIdx.x, lane = tid & 31, wv = tid >> 5;
  int m0 = blockIdx.x * 128 + wv * 16;
  int n0 = blockIdx.y * 128;

  issue_b_chunk(bsh[0], W, n0, 0, tid);          // prologue: chunk 0 in flight

  v8f acc[8] = {};
  for (int k0 = 0; k0 < HID; k0 += 32) {
    int buf = (k0 >> 5) & 1;
    if (k0 + 32 < HID) {
      issue_b_chunk(bsh[buf ^ 1], W, n0, k0 + 32, tid);  // next chunk in flight
      S_WAIT_ASYNC(2);                                   // current chunk landed
    } else {
      S_WAIT_ASYNC(0);
    }
    v16bf a;
    if (AF32) a = load_a_f32((const float*)Aptr, HID, m0, k0, lane);
    else      a = load_a_bf16((const __bf16*)Aptr, HID, m0, k0, lane);
    __syncthreads();                               // chunk visible to all waves
#pragma unroll
    for (int nt = 0; nt < 8; ++nt) {
      v16bf b = load_b_lds(bsh[buf], nt, lane);
      acc[nt] = wmma_bf16(a, b, acc[nt]);
    }
    __syncthreads();                               // done reading this buffer
  }

  int g = lane >> 4;
#pragma unroll
  for (int nt = 0; nt < 8; ++nt) {
    int col = n0 + nt * 16 + (lane & 15);
    float bv = bias1[col];
    if (bias2) bv += bias2[col];
#pragma unroll
    for (int j = 0; j < 8; ++j) {
      int row = m0 + j + 8 * g;
      float v = acc[nt][j] + bv;
      if (OUTF32) ((float*)outp)[(size_t)row * nld + col] = v;
      else        ((__bf16*)outp)[(size_t)row * nld + col] = (__bf16)v;
    }
  }
}

// ---------------------------------------------------- cross-WG barrier ------
static __device__ inline void grid_barrier(int* cnt, int* gen) {
  __syncthreads();
  if (threadIdx.x == 0) {
    __threadfence();
    int g = __hip_atomic_load(gen, __ATOMIC_RELAXED, __HIP_MEMORY_SCOPE_AGENT);
    int arrived = __hip_atomic_fetch_add(cnt, 1, __ATOMIC_ACQ_REL, __HIP_MEMORY_SCOPE_AGENT);
    if (arrived == NWG_DIR - 1) {
      __hip_atomic_store(cnt, 0, __ATOMIC_RELAXED, __HIP_MEMORY_SCOPE_AGENT);
      __hip_atomic_fetch_add(gen, 1, __ATOMIC_RELEASE, __HIP_MEMORY_SCOPE_AGENT);
    } else {
      while (__hip_atomic_load(gen, __ATOMIC_ACQUIRE, __HIP_MEMORY_SCOPE_AGENT) == g)
        __builtin_amdgcn_s_sleep(2);
    }
  }
  __syncthreads();
}

// ------------------------------------------------- persistent LSTM kernel ---
// LDS: Whh slice (128KB, loaded ONCE - time-invariant) + h (64KB/step) + gates
__global__ __launch_bounds__(256) void lstm_rec_kernel(
    const float* __restrict__ xg_f, const float* __restrict__ xg_b,
    const __bf16* __restrict__ Whh_f, const __bf16* __restrict__ Whh_b,
    const float* __restrict__ c0_f, const float* __restrict__ c0_b,
    float* __restrict__ h_all_f, float* __restrict__ h_all_b,
    __bf16* __restrict__ hbuf_f, __bf16* __restrict__ hbuf_b,
    int* __restrict__ bar) {
  extern __shared__ char smem[];
  char*   bsh = smem;                                  // 128*1024 = 128 KB
  __bf16* hsh = (__bf16*)(smem + 131072);              // 64*512 bf16 = 64 KB
  float*  gsh = (float*)(smem + 131072 + 65536);       // 64*128 f32  = 32 KB

  int d = blockIdx.x >> 4;
  int w = blockIdx.x & 15;
  const float*  xg    = d ? xg_b   : xg_f;
  const __bf16* Whh   = d ? Whh_b  : Whh_f;
  const float*  c0    = d ? c0_b   : c0_f;
  float*        h_all = d ? h_all_b : h_all_f;
  __bf16*       hbuf  = d ? hbuf_b : hbuf_f;           // [2][64*512] ping-pong
  int* cnt = bar + 2 * d;
  int* gen = bar + 2 * d + 1;

  int tid  = threadIdx.x;
  int lane = tid & 31;
  int wv   = tid >> 5;            // wave id 0..7
  int mt0  = (wv >> 2) * 2;       // wave's two M-tiles: mt0, mt0+1
  int nt0  = (wv & 3) * 2;        // wave's two N-tiles: nt0, nt0+1

  int hc = tid & 31;              // column within this WG's 32-col H slice
  int rg = tid >> 5;              // row group: rows rg, rg+8, ..., rg+56
  int colH = HSLICE * w + hc;

  // ---- stage this WG's Whh slice into LDS once (time-invariant B matrix) ---
  // bsh row r (0..127) <-> global gate column 512*(r>>5) + 32*w + (r&31)
#pragma unroll
  for (int i = 0; i < 32; ++i) {
    int p = tid + 256 * i;                 // 8192 pieces of 16B
    int row = p >> 6, sub = p & 63;        // 64 x 16B per row (1024B pitch)
    int gcol = 512 * (row >> 5) + HSLICE * w + (row & 31);
    const char* g = (const char*)(Whh + (size_t)gcol * HID) + sub * 16;
    async_g2l_b128(g, bsh + (size_t)row * 1024 + sub * 16);
  }

  float c[8];
#pragma unroll
  for (int e = 0; e < 8; ++e) c[e] = c0[(size_t)(rg + 8 * e) * HID + colH];

  S_WAIT_ASYNC(0);
  __syncthreads();

  for (int t = 0; t < TSEQ; ++t) {
    // prefetch this step's xg gates into registers (completes under GEMM)
    float xr[8][4];
#pragma unroll
    for (int e = 0; e < 8; ++e) {
      size_t xrow = ((size_t)(rg + 8 * e) * TSEQ + t) * GATES;
#pragma unroll
      for (int q = 0; q < 4; ++q) xr[e][q] = xg[xrow + q * HID + colH];
    }

    // async-stage previous h (bf16, 64 KB) into LDS: 16 x 16B per thread
    const char* hp = (const char*)(hbuf + (size_t)(t & 1) * (BATCH * HID));
    char* hs = (char*)hsh;
#pragma unroll
    for (int i = 0; i < 16; ++i) {
      int p = tid + 256 * i;
      async_g2l_b128(hp + (size_t)p * 16, hs + (size_t)p * 16);
    }
    S_WAIT_ASYNC(0);
    __syncthreads();

    // GEMM: 2 M-tiles x 2 N-tiles per wave; A and B fragments each reused 2x
    v8f acc00 = {}, acc01 = {}, acc10 = {}, acc11 = {};
    for (int k0 = 0; k0 < HID; k0 += 32) {
      v16bf a0 = load_a_bf16(hsh, HID, (mt0 + 0) * 16, k0, lane);
      v16bf a1 = load_a_bf16(hsh, HID, (mt0 + 1) * 16, k0, lane);
      v16bf b0 = load_b_lds_full(bsh, nt0 + 0, k0, lane);
      v16bf b1 = load_b_lds_full(bsh, nt0 + 1, k0, lane);
      acc00 = wmma_bf16(a0, b0, acc00);
      acc01 = wmma_bf16(a0, b1, acc01);
      acc10 = wmma_bf16(a1, b0, acc10);
      acc11 = wmma_bf16(a1, b1, acc11);
    }
    __syncthreads();

    // scatter D tiles to gate buffer gsh[64][128]
    int gl = lane >> 4;
#pragma unroll
    for (int j = 0; j < 8; ++j) {
      int c0r = (mt0 + 0) * 16 + j + 8 * gl;
      int c1r = (mt0 + 1) * 16 + j + 8 * gl;
      int cc0 = (nt0 + 0) * 16 + (lane & 15);
      int cc1 = (nt0 + 1) * 16 + (lane & 15);
      gsh[c0r * 128 + cc0] = acc00[j];
      gsh[c0r * 128 + cc1] = acc01[j];
      gsh[c1r * 128 + cc0] = acc10[j];
      gsh[c1r * 128 + cc1] = acc11[j];
    }
    __syncthreads();

    // fused gate math; c lives in registers across all 512 steps
    __bf16* hnext = hbuf + (size_t)((t + 1) & 1) * (BATCH * HID);
#pragma unroll
    for (int e = 0; e < 8; ++e) {
      int r = rg + 8 * e;
      float gi = gsh[r * 128 +  0 + hc] + xr[e][0];
      float gf = gsh[r * 128 + 32 + hc] + xr[e][1];
      float gg = gsh[r * 128 + 64 + hc] + xr[e][2];
      float go = gsh[r * 128 + 96 + hc] + xr[e][3];
      float cn = sigf(gf) * c[e] + sigf(gi) * tanhf(gg);
      float h  = sigf(go) * tanhf(cn);
      c[e] = cn;
      h_all[((size_t)r * TSEQ + t) * HID + colH] = h;
      hnext[(size_t)r * HID + colH] = (__bf16)h;
    }
    grid_barrier(cnt, gen);   // make h visible to all 16 WGs of this direction
  }
}

// ------------------------------------------------------------------ host ----
extern "C" void kernel_launch(void* const* d_in, const int* in_sizes, int n_in,
                              void* d_out, int out_size, void* d_ws, size_t ws_size,
                              hipStream_t stream) {
  (void)in_sizes; (void)n_in; (void)out_size; (void)ws_size;
  const float* x_f   = (const float*)d_in[0];
  const float* x_b   = (const float*)d_in[1];
  const float* h0_f  = (const float*)d_in[2];
  const float* c0_f  = (const float*)d_in[3];
  const float* h0_b  = (const float*)d_in[4];
  const float* c0_b  = (const float*)d_in[5];
  const float* Wih_f = (const float*)d_in[6];
  const float* Whh_f = (const float*)d_in[7];
  const float* bih_f = (const float*)d_in[8];
  const float* bhh_f = (const float*)d_in[9];
  const float* Wih_b = (const float*)d_in[10];
  const float* Whh_b = (const float*)d_in[11];
  const float* bih_b = (const float*)d_in[12];
  const float* bhh_b = (const float*)d_in[13];
  const float* W1    = (const float*)d_in[14];
  const float* b1    = (const float*)d_in[15];
  const float* W2    = (const float*)d_in[16];
  const float* b2    = (const float*)d_in[17];

  char* ws = (char*)d_ws;
  size_t off = 0;
  auto take = [&](size_t bytes) {
    char* p = ws + off;
    off += (bytes + 255) & ~(size_t)255;
    return (void*)p;
  };

  const size_t MT = (size_t)BATCH * TSEQ;                 // 32768 rows
  __bf16* wih_f_bf = (__bf16*)take((size_t)GATES * HID * 2);
  __bf16* wih_b_bf = (__bf16*)take((size_t)GATES * HID * 2);
  __bf16* whh_f_bf = (__bf16*)take((size_t)GATES * HID * 2);
  __bf16* whh_b_bf = (__bf16*)take((size_t)GATES * HID * 2);
  __bf16* w1_bf    = (__bf16*)take((size_t)HID * HID * 2);
  __bf16* w2_bf    = (__bf16*)take((size_t)HID * HID * 2);
  float*  xg_f     = (float*)take(MT * GATES * 4);
  float*  xg_b     = (float*)take(MT * GATES * 4);
  float*  h_all_f  = (float*)take(MT * HID * 4);
  float*  h_all_b  = (float*)take(MT * HID * 4);
  __bf16* hsum_bf  = (__bf16*)take(MT * HID * 2);
  __bf16* p_bf     = (__bf16*)take(MT * HID * 2);
  __bf16* hbuf_f   = (__bf16*)take((size_t)2 * BATCH * HID * 2);
  __bf16* hbuf_b   = (__bf16*)take((size_t)2 * BATCH * HID * 2);
  int*    bar      = (int*)take(4 * sizeof(int));

  // barrier state must be zeroed every call (ws is poisoned, not restored)
  zero_kernel<<<1, 32, 0, stream>>>(bar, 4);

  int nw = GATES * HID;
  f2bf_kernel<<<(nw + 255) / 256, 256, 0, stream>>>(Wih_f, wih_f_bf, nw);
  f2bf_kernel<<<(nw + 255) / 256, 256, 0, stream>>>(Wih_b, wih_b_bf, nw);
  f2bf_kernel<<<(nw + 255) / 256, 256, 0, stream>>>(Whh_f, whh_f_bf, nw);
  f2bf_kernel<<<(nw + 255) / 256, 256, 0, stream>>>(Whh_b, whh_b_bf, nw);
  int nl = HID * HID;
  f2bf_kernel<<<(nl + 255) / 256, 256, 0, stream>>>(W1, w1_bf, nl);
  f2bf_kernel<<<(nl + 255) / 256, 256, 0, stream>>>(W2, w2_bf, nl);
  int nh = BATCH * HID;
  f2bf_kernel<<<(nh + 255) / 256, 256, 0, stream>>>(h0_f, hbuf_f, nh);  // ping buf 0
  f2bf_kernel<<<(nh + 255) / 256, 256, 0, stream>>>(h0_b, hbuf_b, nh);

  // input projections: [32768,512] x [512,2048], A fp32 on the fly, out fp32
  dim3 gx(MT / 128, GATES / 128);
  gemm_kernel<1, 1><<<gx, 256, 0, stream>>>(x_f, wih_f_bf, bih_f, bhh_f, xg_f, GATES);
  gemm_kernel<1, 1><<<gx, 256, 0, stream>>>(x_b, wih_b_bf, bih_b, bhh_b, xg_b, GATES);

  // persistent recurrence: 32 WGs, 224 KB dynamic LDS each (B + h + gates)
  size_t lds = 131072 + (size_t)BATCH * HID * 2 + (size_t)BATCH * 128 * 4;
  lstm_rec_kernel<<<2 * NWG_DIR, 256, lds, stream>>>(
      xg_f, xg_b, whh_f_bf, whh_b_bf, c0_f, c0_b,
      h_all_f, h_all_b, hbuf_f, hbuf_b, bar);

  // out = h_f + h_b (bf16), then two WMMA linears
  int ns = (int)(MT * HID);
  hsum_kernel<<<(ns + 255) / 256, 256, 0, stream>>>(h_all_f, h_all_b, hsum_bf, ns);

  dim3 gl(MT / 128, HID / 128);
  gemm_kernel<0, 0><<<gl, 256, 0, stream>>>(hsum_bf, w1_bf, b1, nullptr, (void*)p_bf, HID);
  gemm_kernel<0, 1><<<gl, 256, 0, stream>>>(p_bf, w2_bf, b2, nullptr, d_out, HID);
}
